// ForecasterBlock_72018011619691
// MI455X (gfx1250) — matrix operand order repaired and checked by reading, not compiled, not run on hardware
//
#include <hip/hip_runtime.h>

// ---------------------------------------------------------------------------
// ForecasterBlock for MI455X (gfx1250, wave32, bf16 WMMA, f32 accumulate).
// All GEMM operands pre-converted to bf16 and pre-swizzled into WMMA fragment
// order: every fragment load is 2x b128 per lane. Hot GEMM loops are manually
// unrolled x2 with ping-pong register stages (loads write directly into the
// loop-carried fragment registers -> no rotation moves, no WMMA->VALU hazard
// NOPs) and dual accumulators per wave sharing one A fragment.
//
// Fragment-swizzled layout for a K x N matrix ("tile" = 32x16 elements):
//   tile(kb, cb) base = ((kb * N/16 + cb) * 32) * 16 ushorts   (1KB per tile)
// A-matrix swizzle (16 rows x K): tile(rb, kb); lane = m + 16*((k>>3)&1),
//   elem = ((k>>4)&1)*8 + (k&7)  -- per ISA 7.12.2 A 16x32 layout.
//
// Workspace (bytes), ~357MB total (see kernel_launch for offsets).
// ---------------------------------------------------------------------------

#define TT  256
#define FF  256
#define HH  512
#define H4  2048
#define NHD 4
#define NKVH 2
#define HDD 128

typedef __attribute__((ext_vector_type(16))) __bf16 v16bf;
typedef __attribute__((ext_vector_type(8)))  float  v8f;
typedef unsigned short ushort_t;

__device__ __forceinline__ ushort_t f2bf(float x) {
  unsigned u = __float_as_uint(x);
  u += 0x7FFFu + ((u >> 16) & 1u);           // round-to-nearest-even
  return (ushort_t)(u >> 16);
}

__device__ __forceinline__ v8f wmma_bf16(v16bf a, v16bf b, v8f c) {
  return __builtin_amdgcn_wmma_f32_16x16x32_bf16(
      false, a, false, b, (short)0, c, false, false);
}

// load one fragment (16 bf16/lane, 32B contiguous) from a swizzled 32x16 tile
__device__ __forceinline__ v16bf frag_ld(const ushort_t* tile) {
  int lane = threadIdx.x & 31;
  union { v16bf v; uint4 q[2]; } f;
  const uint4* p = (const uint4*)(tile + lane * 16);
  f.q[0] = p[0]; f.q[1] = p[1];
  return f.v;
}

// C/D 16x16 f32: vgpr r -> M = r + 8*(lane>>4), N = lane&15
__device__ __forceinline__ void store_c_f32(float* C, int ldc, v8f c) {
  int l = threadIdx.x & 31; int n = l & 15; int mh = l >> 4;
#pragma unroll
  for (int r = 0; r < 8; ++r) C[(size_t)(r + 8 * mh) * ldc + n] = c[r];
}

// swizzled offset for A-matrix element (row m in 16-row block rb, col k)
__device__ __forceinline__ size_t a_swz(int rb, int nkb, int kb, int m, int kw) {
  int lane = m + 16 * ((kw >> 3) & 1);
  int elem = ((kw >> 4) & 1) * 8 + (kw & 7);
  return (((size_t)rb * nkb + kb) * 32 + lane) * 16 + elem;
}

__device__ __forceinline__ float wred32(float v) {
#pragma unroll
  for (int m = 16; m > 0; m >>= 1) v += __shfl_xor(v, m, 32);
  return v;
}

// K-loop over NKB 32-deep k-blocks, unrolled x2 (ping-pong stages), two
// output tiles (B0, B1) sharing one A fragment. astep/bstep in ushorts.
template <int NKB>
__device__ __forceinline__ void gemm2(const ushort_t* __restrict__ Ab, size_t astep,
                                      const ushort_t* __restrict__ B0,
                                      const ushort_t* __restrict__ B1, size_t bstep,
                                      v8f& acc0, v8f& acc1) {
  v16bf a0 = frag_ld(Ab);
  v16bf b0 = frag_ld(B0);
  v16bf b1 = frag_ld(B1);
#pragma unroll 1
  for (int kb = 0; kb < NKB - 2; kb += 2) {
    v16bf a1 = frag_ld(Ab + (size_t)(kb + 1) * astep);
    v16bf b2 = frag_ld(B0 + (size_t)(kb + 1) * bstep);
    v16bf b3 = frag_ld(B1 + (size_t)(kb + 1) * bstep);
    acc0 = wmma_bf16(a0, b0, acc0);
    acc1 = wmma_bf16(a0, b1, acc1);
    a0 = frag_ld(Ab + (size_t)(kb + 2) * astep);
    b0 = frag_ld(B0 + (size_t)(kb + 2) * bstep);
    b1 = frag_ld(B1 + (size_t)(kb + 2) * bstep);
    acc0 = wmma_bf16(a1, b2, acc0);
    acc1 = wmma_bf16(a1, b3, acc1);
  }
  v16bf a1 = frag_ld(Ab + (size_t)(NKB - 1) * astep);
  v16bf b2 = frag_ld(B0 + (size_t)(NKB - 1) * bstep);
  v16bf b3 = frag_ld(B1 + (size_t)(NKB - 1) * bstep);
  acc0 = wmma_bf16(a0, b0, acc0);
  acc1 = wmma_bf16(a0, b1, acc1);
  acc0 = wmma_bf16(a1, b2, acc0);
  acc1 = wmma_bf16(a1, b3, acc1);
}

// ---------------- prep: convert f32 K x N weight into swizzled bf16 ---------
__global__ void __launch_bounds__(256) k_prep(const float* __restrict__ B,
                                              ushort_t* __restrict__ out,
                                              int K, int N) {
  int o = blockIdx.x * 256 + threadIdx.x;   // output linear index (exact grids)
  int i = o & 15;
  int lane = (o >> 4) & 31;
  int rest = o >> 9;
  int ncb = N >> 4;
  int cb = rest % ncb;
  int kb = rest / ncb;
  int n = cb * 16 + (lane & 15);
  int k = kb * 32 + (i >> 3) * 16 + (lane >> 4) * 8 + (i & 7);
  out[o] = f2bf(B[(size_t)k * N + n]);
}

// ---------------- K1: xz = x^T @ Wi + b  (per-sequence transposed GEMM) -----
// grid(128, 128): x = stile(16) * jchunk(8); block covers 16 rows x 256 cols.
__global__ void __launch_bounds__(256) k_xz(const float* __restrict__ series,
                                            const ushort_t* __restrict__ WiS,
                                            const float* __restrict__ bias,
                                            float* __restrict__ xz) {
  __shared__ alignas(16) ushort_t As[8 * 32 * 16];   // 8 kb x 32 x 16 (K=256)
  int n = blockIdx.y;
  int stile = blockIdx.x >> 3;
  int jchunk = blockIdx.x & 7;
  int s0 = stile * 16;
  const float* S = series + (size_t)n * TT * FF;
  for (int idx = threadIdx.x; idx < 16 * FF; idx += 256) {
    int sl = idx & 15; int i = idx >> 4;               // A[sl, i] = S[i, s0+sl]
    As[a_swz(0, 8, i >> 5, sl, i & 31)] = f2bf(S[(size_t)i * FF + s0 + sl]);
  }
  __syncthreads();
  int w = threadIdx.x >> 5;
  int cb0 = jchunk * 16 + w * 2;                       // two 16-col blocks
  int nl = threadIdx.x & 15;
  float bv0 = bias[cb0 * 16 + nl];
  float bv1 = bias[cb0 * 16 + 16 + nl];
  v8f acc0, acc1;
#pragma unroll
  for (int r = 0; r < 8; ++r) { acc0[r] = bv0; acc1[r] = bv1; }
  gemm2<8>(As, 512, WiS + (size_t)cb0 * 512, WiS + (size_t)(cb0 + 1) * 512,
           (size_t)128 * 512, acc0, acc1);
  float* Cp = xz + ((size_t)n * TT + s0) * H4;
  store_c_f32(Cp + cb0 * 16, H4, acc0);
  store_c_f32(Cp + cb0 * 16 + 16, H4, acc1);
}

// ---------------- zero spin-barrier counters --------------------------------
__global__ void k_zero(int* __restrict__ p, int n) {
  for (int i = blockIdx.x * blockDim.x + threadIdx.x; i < n; i += gridDim.x * blockDim.x)
    p[i] = 0;
}

// ---------------- K2: LSTM scan ---------------------------------------------
// grid (8 jb, 8 group): block owns 16 seqs x 64 h-cols (all 4 gate slices).
// h exchanged through hxch in fragment-swizzled form (2x global_load_b128
// A-fragments straight from L2). Dual accumulators per wave + x2 unroll.
__global__ void __launch_bounds__(256) k_lstm(const float* __restrict__ xz,
                                              const ushort_t* __restrict__ WhS,
                                              float* __restrict__ hid,
                                              ushort_t* __restrict__ hxch,
                                              int* __restrict__ cnt) {
  __shared__ float zbuf[16][256 + 8];
  __shared__ float cst[16][64];
  int jb = blockIdx.x;
  int g  = blockIdx.y;
  int w  = threadIdx.x >> 5;
  int l  = threadIdx.x & 31; int nl = l & 15; int mh = l >> 4;

  // wave w owns tiles tt = 2w, 2w+1 -> same gate, adjacent col blocks
  int gate = (2 * w) >> 2, wq0 = (2 * w) & 3;
  int cb0 = gate * 32 + jb * 4 + wq0;                  // of 128 col blocks
  int lc0 = gate * 64 + wq0 * 16;
  const ushort_t* Wb0 = WhS + (size_t)cb0 * 512;
  const ushort_t* Wb1 = WhS + (size_t)(cb0 + 1) * 512;

  for (int idx = threadIdx.x; idx < 16 * 64; idx += 256) (&cst[0][0])[idx] = 0.f;
  __syncthreads();

  for (int s = 0; s < TT; ++s) {
    const ushort_t* hx = hxch + (size_t)(((s - 1) & 1) * 8 + g) * 16 * 512;
    v8f acc0, acc1;
#pragma unroll
    for (int r = 0; r < 8; ++r) {                      // C init = xz[:, s, :]
      int m = r + 8 * mh; int seq = g * 16 + m;
      const float* xp = xz + ((size_t)seq * TT + s) * H4 + cb0 * 16 + nl;
      acc0[r] = xp[0];
      acc1[r] = xp[16];
      if (s + 1 < TT) __builtin_prefetch(xp + H4, 0, 0);   // next step's xz row
    }
    if (s > 0) gemm2<16>(hx, 512, Wb0, Wb1, (size_t)128 * 512, acc0, acc1);
#pragma unroll
    for (int r = 0; r < 8; ++r) {
      zbuf[r + 8 * mh][lc0 + nl] = acc0[r];
      zbuf[r + 8 * mh][lc0 + 16 + nl] = acc1[r];
    }
    __syncthreads();
    // gates i,f,g,o; write h to hid (f32) and swizzled exchange (bf16)
    ushort_t* hw = hxch + (size_t)((s & 1) * 8 + g) * 16 * 512;
    for (int idx = threadIdx.x; idx < 1024; idx += 256) {
      int m = idx >> 6, lc = idx & 63;
      float iv = 1.f / (1.f + __expf(-zbuf[m][lc]));
      float fv = 1.f / (1.f + __expf(-zbuf[m][64 + lc]));
      float gv = tanhf(zbuf[m][128 + lc]);
      float ov = 1.f / (1.f + __expf(-zbuf[m][192 + lc]));
      float c = fv * cst[m][lc] + iv * gv;
      float h = ov * tanhf(c);
      cst[m][lc] = c;
      int seq = g * 16 + m; int hc = jb * 64 + lc;
      hid[((size_t)seq * TT + s) * HH + hc] = h;
      if (s != TT - 1) hw[a_swz(0, 16, hc >> 5, m, hc & 31)] = f2bf(h);
    }
    if (s != TT - 1) {
      __threadfence();
      __syncthreads();
      int* c8 = cnt + g * TT + s;
      if (threadIdx.x == 0) {
        atomicAdd(c8, 1);
        while (__atomic_load_n(c8, __ATOMIC_RELAXED) < 8) __builtin_amdgcn_s_sleep(1);
      }
      __syncthreads();
      __threadfence();
    }
  }
}

// ---------------- K3a: layernorm -> swizzled bf16 normed (attention order) --
__global__ void __launch_bounds__(256) k_ln(const float* __restrict__ hid,
                                            const float* __restrict__ sc,
                                            const float* __restrict__ bi,
                                            ushort_t* __restrict__ nrmS) {
  int r = blockIdx.x * 8 + (threadIdx.x >> 5);          // r_att = (b*256+t)*8+d
  int l = threadIdx.x & 31;
  int b = r >> 11; int rem = r & 2047; int t = rem >> 3; int d = rem & 7;
  const float* src = hid + ((size_t)((b * 8 + d) * TT + t)) * HH;
  float s1 = 0.f, s2 = 0.f;
#pragma unroll
  for (int j = 0; j < 16; ++j) { float x = src[l + 32 * j]; s1 += x; s2 += x * x; }
  s1 = wred32(s1); s2 = wred32(s2);
  float mu = s1 * (1.f / HH);
  float var = s2 * (1.f / HH) - mu * mu;
  float rs = rsqrtf(var + 1e-6f);
  int rb = r >> 4, m = r & 15;
#pragma unroll
  for (int j = 0; j < 16; ++j) {
    int k = l + 32 * j;
    float y = (src[k] - mu) * rs * sc[k] + bi[k];
    nrmS[a_swz(rb, 16, k >> 5, m, k & 31)] = f2bf(y);
  }
}

// ---------------- generic GEMM: C = Aswz @ Bswz, K = 512 --------------------
// grid(x = ncb/16, y = Mrows/16); each wave: 2 adjacent col blocks.
__global__ void __launch_bounds__(256) k_gemm512(const ushort_t* __restrict__ A,
                                                 const ushort_t* __restrict__ B,
                                                 float* __restrict__ C,
                                                 int ncb, int ldc) {
  int w = threadIdx.x >> 5;
  int rb = blockIdx.y;
  int cb0 = blockIdx.x * 16 + w * 2;
  const ushort_t* Ab = A + (size_t)rb * 16 * 512;
  v8f acc0 = {0.f, 0.f, 0.f, 0.f, 0.f, 0.f, 0.f, 0.f};
  v8f acc1 = {0.f, 0.f, 0.f, 0.f, 0.f, 0.f, 0.f, 0.f};
  gemm2<16>(Ab, 512, B + (size_t)cb0 * 512, B + (size_t)(cb0 + 1) * 512,
            (size_t)ncb * 512, acc0, acc1);
  float* Cp = C + (size_t)rb * 16 * ldc;
  store_c_f32(Cp + cb0 * 16, ldc, acc0);
  store_c_f32(Cp + cb0 * 16 + 16, ldc, acc1);
}

// ---------------- K3c: qk-norm in place (length-128 vectors) ----------------
__global__ void __launch_bounds__(256) k_vecnorm128(float* __restrict__ X,
                                                    const float* __restrict__ sc) {
  size_t vid = (size_t)blockIdx.x * 8 + (threadIdx.x >> 5);
  int l = threadIdx.x & 31;
  float* p = X + vid * HDD;
  float s1 = 0.f, s2 = 0.f;
#pragma unroll
  for (int j = 0; j < 4; ++j) { float x = p[l + 32 * j]; s1 += x; s2 += x * x; }
  s1 = wred32(s1); s2 = wred32(s2);
  float mu = s1 * (1.f / HDD);
  float var = s2 * (1.f / HDD) - mu * mu;
  float rs = rsqrtf(var + 1e-6f);
#pragma unroll
  for (int j = 0; j < 4; ++j) { int e = l + 32 * j; p[e] = (p[e] - mu) * rs * sc[e]; }
}

// ---------------- K4: GQA attention (seq = 8, 2 positions per 16x16 tile) ---
__global__ void __launch_bounds__(256) k_attn(const float* __restrict__ q,
                                              const float* __restrict__ kb,
                                              const float* __restrict__ vb,
                                              ushort_t* __restrict__ aoS) {
  __shared__ float sc[8][16][17];
  __shared__ ushort_t aw[8][16][24];
  int w = threadIdx.x >> 5;
  int l = threadIdx.x & 31; int nl = l & 15; int khi = l >> 4;
  int task = blockIdx.x * 8 + w;
  int p = task >> 2;
  int hh = task & 3;
  int kvh = hh >> 1;
  int ra_m = (p * 2 + (nl >> 3)) * 8 + (nl & 7);
  v8f acc = {0.f, 0.f, 0.f, 0.f, 0.f, 0.f, 0.f, 0.f};
  for (int kk = 0; kk < HDD; kk += 32) {
    union { v16bf v; ushort_t u[16]; } fa, fb;
#pragma unroll
    for (int g2 = 0; g2 < 2; ++g2)
#pragma unroll
      for (int e = 0; e < 8; ++e) {
        int k = kk + g2 * 16 + khi * 8 + e;
        fa.u[g2 * 8 + e] = f2bf(q[(size_t)ra_m * (NHD * HDD) + hh * HDD + k]);
      }
#pragma unroll
    for (int i = 0; i < 16; ++i) {
      int k = kk + (i >> 3) * 16 + khi * 8 + (i & 7);
      fb.u[i] = f2bf(kb[(size_t)ra_m * (NKVH * HDD) + kvh * HDD + k]);
    }
    acc = wmma_bf16(fa.v, fb.v, acc);
  }
  const float sscale = 0.0883883476483184f;             // 1/sqrt(128)
#pragma unroll
  for (int r = 0; r < 8; ++r) {
    int m = r + 8 * khi;
    float v = acc[r] * sscale;
    if ((m >> 3) != (nl >> 3)) v = -1e30f;              // mask cross-position
    sc[w][m][nl] = v;
  }
  asm volatile("s_wait_dscnt 0x0" ::: "memory");
  if (l < 16) {
    float mx = -1e30f;
#pragma unroll
    for (int c2 = 0; c2 < 16; ++c2) mx = fmaxf(mx, sc[w][l][c2]);
    float ex[16]; float sum = 0.f;
#pragma unroll
    for (int c2 = 0; c2 < 16; ++c2) { ex[c2] = __expf(sc[w][l][c2] - mx); sum += ex[c2]; }
    float inv = 1.f / sum;
#pragma unroll
    for (int c2 = 0; c2 < 16; ++c2) aw[w][l][c2] = f2bf(ex[c2] * inv);
  }
  asm volatile("s_wait_dscnt 0x0" ::: "memory");
#pragma unroll 1
  for (int nt = 0; nt < 8; ++nt) {
    union { v16bf v; ushort_t u[16]; } fa, fb;
#pragma unroll
    for (int g2 = 0; g2 < 2; ++g2)
#pragma unroll
      for (int e = 0; e < 8; ++e) {
        int k = g2 * 16 + khi * 8 + e;
        fa.u[g2 * 8 + e] = (k < 16) ? aw[w][nl][k] : (ushort_t)0;
      }
#pragma unroll
    for (int i = 0; i < 16; ++i) {
      int k = (i >> 3) * 16 + khi * 8 + (i & 7);
      ushort_t bvv = 0;
      if (k < 16) {
        int rk = (p * 2 + (k >> 3)) * 8 + (k & 7);
        bvv = f2bf(vb[(size_t)rk * (NKVH * HDD) + kvh * HDD + nt * 16 + nl]);
      }
      fb.u[i] = bvv;
    }
    v8f oacc = {0.f, 0.f, 0.f, 0.f, 0.f, 0.f, 0.f, 0.f};
    oacc = wmma_bf16(fa.v, fb.v, oacc);
#pragma unroll
    for (int r = 0; r < 8; ++r) {
      int m = r + 8 * khi;
      int rm = (p * 2 + (m >> 3)) * 8 + (m & 7);
      int col = hh * HDD + nt * 16 + nl;
      aoS[a_swz(rm >> 4, 16, col >> 5, rm & 15, col & 31)] = f2bf(oacc[r]);
    }
  }
}

// ---------------- K6: belief = (hid + mix) @ Wdown --------------------------
__global__ void __launch_bounds__(256) k_final(const float* __restrict__ hid,
                                               const float* __restrict__ mix,
                                               const ushort_t* __restrict__ WdS,
                                               float* __restrict__ out) {
  __shared__ alignas(16) ushort_t As[16 * 32 * 16];     // 16 kb (K=512)
  size_t r0 = (size_t)blockIdx.x * 16;                  // rows in (b,d,t) order
  for (int idx = threadIdx.x; idx < 16 * HH; idx += 256) {
    int m = idx >> 9, k = idx & 511;
    size_t rh = r0 + m;
    int b = (int)(rh >> 11); int d = (int)((rh >> 8) & 7); int t = (int)(rh & 255);
    size_t ra = ((size_t)(b * TT + t)) * 8 + d;
    As[a_swz(0, 16, k >> 5, m, k & 31)] = f2bf(hid[rh * HH + k] + mix[ra * HH + k]);
  }
  __syncthreads();
  int w = threadIdx.x >> 5;
  int cb0 = w * 2;                                      // 16 col blocks (N=256)
  v8f acc0 = {0.f, 0.f, 0.f, 0.f, 0.f, 0.f, 0.f, 0.f};
  v8f acc1 = {0.f, 0.f, 0.f, 0.f, 0.f, 0.f, 0.f, 0.f};
  gemm2<16>(As, 512, WdS + (size_t)cb0 * 512, WdS + (size_t)(cb0 + 1) * 512,
            (size_t)16 * 512, acc0, acc1);
  store_c_f32(out + r0 * FF + cb0 * 16, FF, acc0);
  store_c_f32(out + r0 * FF + cb0 * 16 + 16, FF, acc1);
}

// ---------------------------------------------------------------------------
extern "C" void kernel_launch(void* const* d_in, const int* in_sizes, int n_in,
                              void* d_out, int out_size, void* d_ws, size_t ws_size,
                              hipStream_t stream) {
  const float* series = (const float*)d_in[0];
  const float* Wi  = (const float*)d_in[1];
  const float* Wh  = (const float*)d_in[2];
  const float* bb  = (const float*)d_in[3];
  const float* lns = (const float*)d_in[4];
  const float* lnb = (const float*)d_in[5];
  const float* Wq  = (const float*)d_in[6];
  const float* Wk  = (const float*)d_in[7];
  const float* Wv  = (const float*)d_in[8];
  const float* qns = (const float*)d_in[9];
  const float* kns = (const float*)d_in[10];
  const float* Wo  = (const float*)d_in[11];
  const float* Wd  = (const float*)d_in[12];

  char* ws = (char*)d_ws;
  float* xz          = (float*)(ws + 0);
  float* qb          = (float*)(ws + 0);                 // overlays xz
  float* kbuf        = (float*)(ws + 67108864);
  float* vbuf        = (float*)(ws + 100663296);
  ushort_t* aoS      = (ushort_t*)(ws + 134217728);
  float* mix         = (float*)(ws + 167772160);
  float* hid         = (float*)(ws + 268435456);
  ushort_t* nrmS     = (ushort_t*)(ws + 335544320);
  ushort_t* hxch     = (ushort_t*)(ws + 369098752);
  int* cnt           = (int*)(ws + 369360896);
  ushort_t* WiS      = (ushort_t*)(ws + 369369088);
  ushort_t* WhS      = (ushort_t*)(ws + 370417664);
  ushort_t* WqS      = (ushort_t*)(ws + 372514816);
  ushort_t* WkS      = (ushort_t*)(ws + 373039104);
  ushort_t* WvS      = (ushort_t*)(ws + 373301248);
  ushort_t* WoS      = (ushort_t*)(ws + 373563392);
  ushort_t* WdS      = (ushort_t*)(ws + 374087680);
  float* out = (float*)d_out;

  // one-time weight convert + swizzle (bf16, fragment order)
  k_prep<<<(FF * H4) / 256, 256, 0, stream>>>(Wi, WiS, FF, H4);
  k_prep<<<(HH * H4) / 256, 256, 0, stream>>>(Wh, WhS, HH, H4);
  k_prep<<<(HH * 512) / 256, 256, 0, stream>>>(Wq, WqS, HH, 512);
  k_prep<<<(HH * 256) / 256, 256, 0, stream>>>(Wk, WkS, HH, 256);
  k_prep<<<(HH * 256) / 256, 256, 0, stream>>>(Wv, WvS, HH, 256);
  k_prep<<<(HH * 512) / 256, 256, 0, stream>>>(Wo, WoS, HH, 512);
  k_prep<<<(HH * 256) / 256, 256, 0, stream>>>(Wd, WdS, HH, 256);

  k_xz  <<<dim3(128, 128), 256, 0, stream>>>(series, WiS, bb, xz);
  k_zero<<<8, 256, 0, stream>>>(cnt, 8 * TT);
  k_lstm<<<dim3(8, 8), 256, 0, stream>>>(xz, WhS, hid, hxch, cnt);
  k_ln  <<<4096, 256, 0, stream>>>(hid, lns, lnb, nrmS);
  k_gemm512<<<dim3(2, 2048), 256, 0, stream>>>(nrmS, WqS, qb,   32, 512);
  k_gemm512<<<dim3(1, 2048), 256, 0, stream>>>(nrmS, WkS, kbuf, 16, 256);
  k_gemm512<<<dim3(1, 2048), 256, 0, stream>>>(nrmS, WvS, vbuf, 16, 256);
  k_vecnorm128<<<16384, 256, 0, stream>>>(qb, qns);
  k_vecnorm128<<<8192, 256, 0, stream>>>(kbuf, kns);
  k_attn<<<1024, 256, 0, stream>>>(qb, kbuf, vbuf, aoS);
  k_gemm512<<<dim3(2, 2048), 256, 0, stream>>>(aoS, WoS, mix, 32, 512);
  k_final<<<2048, 256, 0, stream>>>(hid, mix, WdS, out);
}
